// GCN_4320737100493
// MI455X (gfx1250) — matrix operand order repaired and checked by reading, not compiled, not run on hardware
//
#include <hip/hip_runtime.h>
#include <hip/hip_bf16.h>
#include <math.h>

typedef __attribute__((ext_vector_type(16))) __bf16 bf16x16;
typedef __attribute__((ext_vector_type(8)))  float  f32x8;

#define H_DIM 128

// ---------------- utility ----------------
__global__ void zero_kernel(float* __restrict__ p, int n) {
  int i = blockIdx.x * blockDim.x + threadIdx.x;
  if (i < n) p[i] = 0.f;
}

// ---------------- edge aggregation: agg[dst] += x[src] * em ----------------
__global__ void edge_agg_kernel(const float* __restrict__ x, const int* __restrict__ src,
                                const int* __restrict__ dst, const float* __restrict__ em,
                                float* __restrict__ agg, int nE) {
  int i = blockIdx.x * blockDim.x + threadIdx.x;   // one thread = one edge x 4 feats
  int e = i >> 5;
  if (e >= nE) return;
  int c = (i & 31) << 2;
  float m = em ? em[e] : 1.f;
  if (m == 0.f) return;
  int s = src[e], d = dst[e];
  const float* xs = x + (size_t)s * H_DIM + c;
  float* ad = agg + (size_t)d * H_DIM + c;
  atomicAdd(ad + 0, xs[0] * m);
  atomicAdd(ad + 1, xs[1] * m);
  atomicAdd(ad + 2, xs[2] * m);
  atomicAdd(ad + 3, xs[3] * m);
}

// ---------------- weight pre-swizzle: W[K,N] fp32 -> bf16 B-fragments ----------------
// Fragment (ks, fcol): 32 lanes x 16 bf16, element j<8 -> K 8*half+j, j>=8 -> K 16+8*half+(j-8)
// of the 32-K step, column = fcol*16 + (lane&15).
// Stored contiguously: frag index = (ks * (N/16) + fcol) * 32 + lane, 16 bf16 each.
__global__ void w_prep_kernel(const float* __restrict__ W, __bf16* __restrict__ Wf,
                              int K, int N) {
  int tid = blockIdx.x * blockDim.x + threadIdx.x;
  int nf = N >> 4;
  int total = (K >> 5) * nf * 32;
  if (tid >= total) return;
  int lane = tid & 31;
  int fcol = (tid >> 5) % nf;
  int ks   = tid / (32 * nf);
  int half = lane >> 4, nl = lane & 15;
  int n = fcol * 16 + nl;
  bf16x16 frag;
  for (int j = 0; j < 16; ++j)
    frag[j] = (__bf16)W[(size_t)(ks * 32 + (j < 8 ? half * 8 + j : 16 + half * 8 + (j - 8))) * N + n];
  *(bf16x16*)(Wf + (size_t)tid * 16) = frag;
}

// ---------------- WMMA GEMM: C = act(A[M,K] @ W[K,N] (+bias) (+C)) ----------------
// block = 128 threads (4 waves); wave tile 32(M) x 64(N); block tile 128 x 64.
// B comes pre-swizzled in Wf (contiguous 32B per lane per fragment).
__global__ void __launch_bounds__(128)
wmma_gemm(const float* __restrict__ A, const __bf16* __restrict__ Wf,
          const float* __restrict__ bias, float* __restrict__ C,
          int M, int K, int N, int accum, int relu) {
  const int lane = threadIdx.x & 31;
  const int wave = threadIdx.x >> 5;
  const int half = lane >> 4;                // 0: lanes 0-15, 1: lanes 16-31
  const int l16  = lane & 15;
  const int rowbase = blockIdx.x * 128 + wave * 32;
  const int colbase = blockIdx.y * 64;
  const int nf = N >> 4;                     // col-fragments per K-step in Wf
  const int f0 = colbase >> 4;

  f32x8 acc[2][4];
  for (int rt = 0; rt < 2; ++rt)
    for (int f = 0; f < 4; ++f)
      for (int r = 0; r < 8; ++r) acc[rt][f][r] = 0.f;

  const float* ap0 = A + (size_t)(rowbase + l16) * K;
  const float* ap1 = A + (size_t)(rowbase + 16 + l16) * K;

  const int ksteps = K >> 5;
  for (int ks = 0; ks < ksteps; ++ks) {
    const int k0 = ks << 5;
    // A fragments (fp32 -> bf16 in registers): K chunks {kb..kb+7, kb+16..kb+23}
    bf16x16 a0, a1;
    {
      const float* p0 = ap0 + k0 + half * 8;
      const float* p1 = ap1 + k0 + half * 8;
      for (int j = 0; j < 8; ++j) {
        a0[j] = (__bf16)p0[j];  a0[8 + j] = (__bf16)p0[16 + j];
        a1[j] = (__bf16)p1[j];  a1[8 + j] = (__bf16)p1[16 + j];
      }
      if (k0 + 32 < K) {                     // prefetch next K-chunk of A
        __builtin_prefetch(p0 + 32, 0, 1);
        __builtin_prefetch(p1 + 32, 0, 1);
      }
    }
    // B fragments: one contiguous 32B load per lane, fed to two WMMAs
    const __bf16* wrow = Wf + ((size_t)(ks * nf + f0) * 32 + lane) * 16;
    for (int f = 0; f < 4; ++f) {
      bf16x16 bf = *(const bf16x16*)(wrow + (size_t)f * 32 * 16);
      acc[0][f] = __builtin_amdgcn_wmma_f32_16x16x32_bf16(
          false, a0, false, bf, (short)0, acc[0][f], false, false);
      acc[1][f] = __builtin_amdgcn_wmma_f32_16x16x32_bf16(
          false, a1, false, bf, (short)0, acc[1][f], false, false);
    }
  }

  // epilogue: C/D layout: elem r -> row (+ half*8 + r), col colbase + f*16 + l16
  for (int rt = 0; rt < 2; ++rt)
    for (int f = 0; f < 4; ++f) {
      int ncol = colbase + f * 16 + l16;
      float bv = bias ? bias[ncol] : 0.f;
      for (int r = 0; r < 8; ++r) {
        int mrow = rowbase + rt * 16 + half * 8 + r;
        size_t idx = (size_t)mrow * N + ncol;
        float v = acc[rt][f][r] + bv;
        if (accum) v += C[idx];
        if (relu)  v = fmaxf(v, 0.f);
        C[idx] = v;
      }
    }
}

// ---------------- score = tanh((h . p) / ||p||), one wave per node ----------------
__global__ void score_kernel(const float* __restrict__ h, const float* __restrict__ p,
                             float* __restrict__ score, int nodes) {
  int warp = (blockIdx.x * blockDim.x + threadIdx.x) >> 5;
  int lane = threadIdx.x & 31;
  if (warp >= nodes) return;
  const float* hp = h + (size_t)warp * H_DIM;
  float dot = 0.f, pp = 0.f;
  for (int i = lane; i < H_DIM; i += 32) {
    float pv = p[i];
    dot += hp[i] * pv;
    pp  += pv * pv;
  }
  for (int off = 16; off > 0; off >>= 1) {
    dot += __shfl_xor(dot, off, 32);
    pp  += __shfl_xor(pp,  off, 32);
  }
  if (lane == 0) score[warp] = tanhf(dot * rsqrtf(pp));
}

// ---------------- per-graph top-k via LDS bitonic sort (256 padded) ----------------
__global__ void __launch_bounds__(256)
topk_kernel(const float* __restrict__ score, int n_per, int k,
            int* __restrict__ perm, int* __restrict__ new_id) {
  __shared__ float skey[256];
  __shared__ int   sidx[256];
  int g = blockIdx.x, t = threadIdx.x;
  skey[t] = (t < n_per) ? score[g * n_per + t] : -1e30f;
  sidx[t] = t;
  if (t < n_per) new_id[g * n_per + t] = -1;
  __syncthreads();
  for (int kk = 2; kk <= 256; kk <<= 1)
    for (int j = kk >> 1; j > 0; j >>= 1) {
      int ix = t ^ j;
      if (ix > t) {
        bool desc = ((t & kk) == 0);
        float a = skey[t], b = skey[ix];
        bool sw = desc ? (a < b) : (a > b);
        if (sw) {
          skey[t] = b; skey[ix] = a;
          int tmp = sidx[t]; sidx[t] = sidx[ix]; sidx[ix] = tmp;
        }
      }
      __syncthreads();
    }
  if (t < k) {
    int old = sidx[t];
    perm[g * k + t] = g * n_per + old;
    new_id[g * n_per + old] = g * k + t;
  }
}

// ---------------- new_x[i] = h[perm[i]] * score[perm[i]] ----------------
__global__ void gather_scale_kernel(const float* __restrict__ h, const int* __restrict__ perm,
                                    const float* __restrict__ score, float* __restrict__ out,
                                    int newNodes) {
  int i = blockIdx.x * blockDim.x + threadIdx.x;
  int node = i >> 5;
  if (node >= newNodes) return;
  int c = (i & 31) << 2;
  int p = perm[node];
  float s = score[p];
  const float* hp = h + (size_t)p * H_DIM + c;
  float* op = out + (size_t)node * H_DIM + c;
  op[0] = hp[0] * s; op[1] = hp[1] * s; op[2] = hp[2] * s; op[3] = hp[3] * s;
}

// ---------------- remap edges through new_id, mask dropped endpoints ----------------
__global__ void edge_remap_kernel(const int* __restrict__ src_in, const int* __restrict__ dst_in,
                                  const float* __restrict__ em_in, int* __restrict__ src_out,
                                  int* __restrict__ dst_out, float* __restrict__ em_out,
                                  const int* __restrict__ new_id, int nE) {
  int e = blockIdx.x * blockDim.x + threadIdx.x;
  if (e >= nE) return;
  float m = em_in ? em_in[e] : 1.f;
  int ns = new_id[src_in[e]];
  int nd = new_id[dst_in[e]];
  float valid = (ns >= 0 && nd >= 0) ? m : 0.f;
  src_out[e] = ns > 0 ? ns : 0;
  dst_out[e] = nd > 0 ? nd : 0;
  em_out[e] = valid;
}

// ---------------- readout: z[g] += [max_r h, mean_r h] ----------------
__global__ void readout_kernel(const float* __restrict__ h, float* __restrict__ z, int k) {
  int g = blockIdx.x, t = threadIdx.x;  // 128 threads, one per feature
  const float* hp = h + (size_t)g * k * H_DIM + t;
  float mx = -1e30f, sm = 0.f;
  for (int r = 0; r < k; ++r) {
    float v = hp[(size_t)r * H_DIM];
    mx = fmaxf(mx, v);
    sm += v;
  }
  z[g * 256 + t]       += mx;
  z[g * 256 + 128 + t] += sm / (float)k;
}

// ---------------- head: logits = m2 @ wl3 + bl3; log_softmax ----------------
__global__ void head_kernel(const float* __restrict__ m2, const float* __restrict__ wl3,
                            const float* __restrict__ bl3, float* __restrict__ out) {
  __shared__ float logits[10];
  __shared__ float smax, slse;
  int row = blockIdx.x, t = threadIdx.x;
  if (t < 10) {
    float a = bl3[t];
    const float* zp = m2 + row * 64;
    for (int j = 0; j < 64; ++j) a += zp[j] * wl3[j * 10 + t];
    logits[t] = a;
  }
  __syncthreads();
  if (t == 0) {
    float m = logits[0];
    for (int i = 1; i < 10; ++i) m = fmaxf(m, logits[i]);
    float s = 0.f;
    for (int i = 0; i < 10; ++i) s += expf(logits[i] - m);
    smax = m; slse = logf(s);
  }
  __syncthreads();
  if (t < 10) out[row * 10 + t] = logits[t] - smax - slse;
}

extern "C" void kernel_launch(void* const* d_in, const int* in_sizes, int n_in,
                              void* d_out, int out_size, void* d_ws, size_t ws_size,
                              hipStream_t stream) {
  (void)in_sizes; (void)n_in; (void)out_size; (void)ws_size;
  const float* x       = (const float*)d_in[0];
  const int*   src     = (const int*)  d_in[1];
  const int*   dst     = (const int*)  d_in[2];
  const float* w1_rel  = (const float*)d_in[3];
  const float* w1_root = (const float*)d_in[4];
  const float* b1      = (const float*)d_in[5];
  const float* p1      = (const float*)d_in[6];
  const float* w2_rel  = (const float*)d_in[7];
  const float* w2_root = (const float*)d_in[8];
  const float* b2      = (const float*)d_in[9];
  const float* p2      = (const float*)d_in[10];
  const float* w3_rel  = (const float*)d_in[11];
  const float* w3_root = (const float*)d_in[12];
  const float* b3      = (const float*)d_in[13];
  const float* p3      = (const float*)d_in[14];
  const float* wl1     = (const float*)d_in[15];
  const float* bl1     = (const float*)d_in[16];
  const float* wl2     = (const float*)d_in[17];
  const float* bl2     = (const float*)d_in[18];
  const float* wl3     = (const float*)d_in[19];
  const float* bl3     = (const float*)d_in[20];
  float* out = (float*)d_out;

  const int Bg = 512, N0 = 256, H = 128, E = 512 * 256 * 8;
  const int k1 = 205, k2 = 164, k3 = 132;          // ceil(0.8 * {256,205,164})
  const int n0 = Bg * N0, n1 = Bg * k1, n2 = Bg * k2, n3 = Bg * k3;

  // workspace layout
  char* wsp = (char*)d_ws;
  auto alloc = [&](size_t bytes) -> void* {
    void* p = wsp;
    wsp += (bytes + 255) & ~(size_t)255;
    return p;
  };
  float* agg = (float*)alloc((size_t)n0 * H * 4);
  float* t0  = (float*)alloc((size_t)n0 * H * 4);
  float* t1  = (float*)alloc((size_t)n0 * H * 4);
  float* sc  = (float*)alloc((size_t)n0 * 4);
  float* z   = (float*)alloc((size_t)Bg * 2 * H * 4);
  float* m1  = (float*)alloc((size_t)Bg * H * 4);
  float* m2  = (float*)alloc((size_t)Bg * 64 * 4);
  int* perm  = (int*)alloc((size_t)n0 * 4);
  int* nid   = (int*)alloc((size_t)n0 * 4);
  int* srcc  = (int*)alloc((size_t)E * 4);
  int* dstc  = (int*)alloc((size_t)E * 4);
  float* emc = (float*)alloc((size_t)E * 4);
  __bf16* wfrag = (__bf16*)alloc((size_t)256 * 128 * 2);   // max K*N bf16

  auto zero = [&](float* p, int n) {
    zero_kernel<<<(n + 255) / 256, 256, 0, stream>>>(p, n);
  };
  auto gemm = [&](const float* A, const float* W_, const float* bias, float* Cp,
                  int M, int K, int N, int accum, int relu) {
    int total = (K >> 5) * (N >> 4) * 32;
    w_prep_kernel<<<(total + 255) / 256, 256, 0, stream>>>(W_, wfrag, K, N);
    dim3 grid(M / 128, N / 64);
    wmma_gemm<<<grid, 128, 0, stream>>>(A, wfrag, bias, Cp, M, K, N, accum, relu);
  };
  auto eagg = [&](const float* h, const int* s, const int* d, const float* em, float* ag) {
    int total = E * 32;
    edge_agg_kernel<<<total / 256, 256, 0, stream>>>(h, s, d, em, ag, E);
  };
  auto score = [&](const float* h, const float* p, int nodes) {
    score_kernel<<<(nodes * 32 + 127) / 128, 128, 0, stream>>>(h, p, sc, nodes);
  };
  auto gath = [&](const float* h, float* o, int newNodes) {
    gather_scale_kernel<<<(newNodes * 32 + 255) / 256, 256, 0, stream>>>(h, perm, sc, o, newNodes);
  };

  // ---- layer 1 ----
  zero(agg, n0 * H);
  eagg(x, src, dst, nullptr, agg);
  gemm(agg, w1_rel, b1, t0, n0, H, H, 0, 0);
  gemm(x,   w1_root, nullptr, t0, n0, H, H, 1, 1);
  score(t0, p1, n0);
  topk_kernel<<<Bg, 256, 0, stream>>>(sc, N0, k1, perm, nid);
  gath(t0, t1, n1);
  edge_remap_kernel<<<(E + 255) / 256, 256, 0, stream>>>(src, dst, nullptr,
                                                         srcc, dstc, emc, nid, E);
  zero(z, Bg * 2 * H);
  readout_kernel<<<Bg, 128, 0, stream>>>(t1, z, k1);

  // ---- layer 2 ----
  zero(agg, n1 * H);
  eagg(t1, srcc, dstc, emc, agg);
  gemm(agg, w2_rel, b2, t0, n1, H, H, 0, 0);
  gemm(t1,  w2_root, nullptr, t0, n1, H, H, 1, 1);
  score(t0, p2, n1);
  topk_kernel<<<Bg, 256, 0, stream>>>(sc, k1, k2, perm, nid);
  gath(t0, t1, n2);
  edge_remap_kernel<<<(E + 255) / 256, 256, 0, stream>>>(srcc, dstc, emc,
                                                         srcc, dstc, emc, nid, E);
  readout_kernel<<<Bg, 128, 0, stream>>>(t1, z, k2);

  // ---- layer 3 ----
  zero(agg, n2 * H);
  eagg(t1, srcc, dstc, emc, agg);
  gemm(agg, w3_rel, b3, t0, n2, H, H, 0, 0);
  gemm(t1,  w3_root, nullptr, t0, n2, H, H, 1, 1);
  score(t0, p3, n2);
  topk_kernel<<<Bg, 256, 0, stream>>>(sc, k2, k3, perm, nid);
  gath(t0, t1, n3);
  readout_kernel<<<Bg, 128, 0, stream>>>(t1, z, k3);

  // ---- MLP head ----
  gemm(z,  wl1, bl1, m1, Bg, 2 * H, H, 0, 1);
  gemm(m1, wl2, bl2, m2, Bg, H, 64, 0, 1);
  head_kernel<<<Bg, 32, 0, stream>>>(m2, wl3, bl3, out);
}
